// MambaEncoderSwarmModel_29300266893423
// MI455X (gfx1250) — matrix operand order repaired and checked by reading, not compiled, not run on hardware
//
#include <hip/hip_runtime.h>
#include <hip/hip_bf16.h>

// ---------------------------------------------------------------------------
// MI455X (gfx1250): bf16 WMMA GEMMs (f32 accum, native v_cvt_pk_bf16_f32
// conversion), WMMA flash attention with async global->LDS staging (ASYNCcnt),
// wave32 shfl-based selective scan.
// ---------------------------------------------------------------------------

typedef __bf16 v16bf __attribute__((ext_vector_type(16)));
typedef __bf16 v2bf  __attribute__((ext_vector_type(2)));
typedef float  v8f   __attribute__((ext_vector_type(8)));
typedef int    v4i   __attribute__((ext_vector_type(4)));

union BF16x16 { v16bf v; unsigned u[8]; };

// Native f32->bf16 (RNE); lowers to v_cvt_pk_bf16_f32 on gfx1250.
__device__ __forceinline__ unsigned packbf2(float lo, float hi) {
  v2bf p;
  p.x = (__bf16)lo;
  p.y = (__bf16)hi;
  unsigned u;
  __builtin_memcpy(&u, &p, 4);
  return u;
}
__device__ __forceinline__ float siluf(float x) { return x / (1.f + __expf(-x)); }

#if defined(__gfx1250__) && __has_builtin(__builtin_amdgcn_global_load_async_to_lds_b128) && __has_builtin(__builtin_amdgcn_s_wait_asynccnt)
#define HAVE_ASYNC_LDS 1
typedef __attribute__((address_space(1))) v4i* g_v4i_t;   // global int4*
typedef __attribute__((address_space(3))) v4i* l_v4i_t;   // LDS int4*
#else
#define HAVE_ASYNC_LDS 0
#endif

// ---------------------------------------------------------------------------
// Generic GEMM: C[M,N] = act(A[M,K] @ W[N,K]^T + bias) (+ residual)
// Block tile 128x64, 8 waves, each wave -> 16 rows x 64 cols (4 WMMA accums).
// bf16 operands staged in LDS, f32 accumulation (v_wmma_f32_16x16x32_bf16).
// ---------------------------------------------------------------------------
#define BM 128
#define BN 64
#define BK 32

template<int ACT, bool HAS_BIAS, bool HAS_RES>
__global__ __launch_bounds__(256)
void gemm_bf16_wmma(const float* __restrict__ A, int lda,
                    const float* __restrict__ W,              // N x K, row-major
                    const float* __restrict__ bias,
                    const float* __restrict__ Res, int ldr,
                    float* __restrict__ C, int ldc,
                    int M, int N, int K)
{
  __shared__ unsigned sAu[BM * BK / 2];   // bf16x2 packed
  __shared__ unsigned sBu[BN * BK / 2];

  const int m0   = blockIdx.y * BM;
  const int n0   = blockIdx.x * BN;
  const int tid  = threadIdx.x;
  const int wid  = tid >> 5;
  const int lane = tid & 31;

  v8f acc[4] = {};

  for (int k0 = 0; k0 < K; k0 += BK) {
    // ---- prefetch next K tile (global_prefetch_b8) ----
    if (k0 + BK < K) {
      int pm = m0 + (tid & (BM - 1)); if (pm >= M) pm = M - 1;
      int pn = n0 + (tid & (BN - 1)); if (pn >= N) pn = N - 1;
      __builtin_prefetch(A + (size_t)pm * lda + k0 + BK, 0, 1);
      __builtin_prefetch(W + (size_t)pn * K   + k0 + BK, 0, 1);
    }
    // ---- stage A tile (f32 -> packed bf16x2) ----
    for (int i = tid; i < BM * BK / 2; i += 256) {
      int e = i << 1; int r = e >> 5; int c = e & 31;
      int gm = m0 + r;
      float v0 = (gm < M && (k0 + c)     < K) ? A[(size_t)gm * lda + k0 + c]     : 0.f;
      float v1 = (gm < M && (k0 + c + 1) < K) ? A[(size_t)gm * lda + k0 + c + 1] : 0.f;
      sAu[i] = packbf2(v0, v1);
    }
    // ---- stage B tile (W rows n0..n0+63) ----
    for (int i = tid; i < BN * BK / 2; i += 256) {
      int e = i << 1; int r = e >> 5; int c = e & 31;
      int gn = n0 + r;
      float v0 = (gn < N && (k0 + c)     < K) ? W[(size_t)gn * K + k0 + c]     : 0.f;
      float v1 = (gn < N && (k0 + c + 1) < K) ? W[(size_t)gn * K + k0 + c + 1] : 0.f;
      sBu[i] = packbf2(v0, v1);
    }
    __syncthreads();

    // ---- A fragment (16x32 bf16, ISA 7.12.2 layout) ----
    const int mrow = (wid << 4) + (lane & 15);
    const int kbA  = (lane & 16) >> 1;            // 0 or 8
    BF16x16 af;
    #pragma unroll
    for (int v = 0; v < 8; ++v) {
      int kk = ((v & 4) << 2) + kbA + ((v & 3) << 1);
      af.u[v] = sAu[(mrow * BK + kk) >> 1];
    }
    // ---- 4 B fragments / WMMAs ----
    const int kbB = (lane & 16);                  // 0 or 16
    #pragma unroll
    for (int nt = 0; nt < 4; ++nt) {
      int ncol = (nt << 4) + (lane & 15);
      BF16x16 bf;
      #pragma unroll
      for (int v = 0; v < 8; ++v) {
        int kk = ((v & 4) << 2) + kbA + ((v & 3) << 1);
        bf.u[v] = sBu[(ncol * BK + kk) >> 1];
      }
      acc[nt] = __builtin_amdgcn_wmma_f32_16x16x32_bf16(
          false, af.v, false, bf.v, (short)0, acc[nt], false, false);
    }
    __syncthreads();
  }

  // ---- epilogue: C layout = lane(n), half(lane>=16)->M+8, VGPR r -> M+r ----
  const int mbase = m0 + (wid << 4) + ((lane & 16) >> 1);
  const int ncl   = lane & 15;
  #pragma unroll
  for (int nt = 0; nt < 4; ++nt) {
    int gn = n0 + (nt << 4) + ncl;
    if (gn >= N) continue;
    float b = HAS_BIAS ? bias[gn] : 0.f;
    #pragma unroll
    for (int r = 0; r < 8; ++r) {
      int gm = mbase + r;
      if (gm >= M) continue;
      float v = acc[nt][r] + b;
      if (ACT == 1) v = siluf(v);
      if (ACT == 2) v = (v > 20.f) ? v : log1pf(__expf(v));   // softplus
      if (HAS_RES)  v += Res[(size_t)gm * ldr + gn];
      C[(size_t)gm * ldc + gn] = v;
    }
  }
}

// ---------------------------------------------------------------------------
// Flash attention: 1 wave per (head, 16-query tile). hd=96, keys tiled by 32.
// K/V tiles staged via async global->LDS (ASYNCcnt) when available.
// scores: 6 WMMAs/tile, P@V: 6 WMMAs/tile, online softmax via shfl_xor(16).
// ---------------------------------------------------------------------------
__global__ __launch_bounds__(32)
void attn_flash(const float* __restrict__ Q,   // SQ x 768
                const float* __restrict__ Kb,  // SK x 768
                const float* __restrict__ Vb,  // SK x 768
                float* __restrict__ O,         // SQ x 768
                int SQ, int SK)
{
  __shared__ float sK[32 * 96];
  __shared__ float sV[32 * 96];
  __shared__ float sP[16 * 32];

  const int lane = threadIdx.x;
  const int q0   = blockIdx.x << 4;
  const int hoff = blockIdx.y * 96;
  const float scale = 0.1020620726f;            // 1/sqrt(96)

  // Q fragments (A-matrix, 3 chunks of K=32 over head dim), scale folded in.
  const int m   = lane & 15;
  const int kbA = (lane & 16) >> 1;
  const int kbB = (lane & 16);
  BF16x16 qf[3];
  {
    const float* qr = Q + (size_t)(q0 + m) * 768 + hoff;
    #pragma unroll
    for (int c = 0; c < 3; ++c)
      #pragma unroll
      for (int v = 0; v < 8; ++v) {
        int kk = c * 32 + ((v & 4) << 2) + kbA + ((v & 3) << 1);
        qf[c].u[v] = packbf2(qr[kk] * scale, qr[kk + 1] * scale);
      }
  }

  float mst[8], lst[8];
  #pragma unroll
  for (int r = 0; r < 8; ++r) { mst[r] = -1e30f; lst[r] = 0.f; }
  v8f oacc[6] = {};

  for (int kt = 0; kt < SK; kt += 32) {
#if HAVE_ASYNC_LDS
    // 16B per lane per issue; 24 issues/lane cover both 32x96 f32 tiles.
    for (int ch = lane; ch < 768; ch += 32) {
      int r = ch / 24;
      int c = (ch - r * 24) << 2;
      __builtin_amdgcn_global_load_async_to_lds_b128(
          (g_v4i_t)(Kb + (size_t)(kt + r) * 768 + hoff + c),
          (l_v4i_t)&sK[r * 96 + c], 0, 0);
      __builtin_amdgcn_global_load_async_to_lds_b128(
          (g_v4i_t)(Vb + (size_t)(kt + r) * 768 + hoff + c),
          (l_v4i_t)&sV[r * 96 + c], 0, 0);
    }
    __builtin_amdgcn_s_wait_asynccnt(0);
    __syncthreads();
#else
    for (int i = lane; i < 32 * 96; i += 32) {
      int r = i / 96, c = i % 96;
      sK[i] = Kb[(size_t)(kt + r) * 768 + hoff + c];
      sV[i] = Vb[(size_t)(kt + r) * 768 + hoff + c];
    }
    __syncthreads();
#endif

    // scores S (16 x 32): two 16x16 accumulators
    v8f s[2] = {};
    #pragma unroll
    for (int j = 0; j < 2; ++j)
      #pragma unroll
      for (int c = 0; c < 3; ++c) {
        int key = (j << 4) + (lane & 15);
        BF16x16 kf;
        #pragma unroll
        for (int v = 0; v < 8; ++v) {
          int kk = c * 32 + ((v & 4) << 2) + kbA + ((v & 3) << 1);
          kf.u[v] = packbf2(sK[key * 96 + kk], sK[key * 96 + kk + 1]);
        }
        s[j] = __builtin_amdgcn_wmma_f32_16x16x32_bf16(
            false, qf[c].v, false, kf.v, (short)0, s[j], false, false);
      }

    // online softmax (rows live in 16-lane halves)
    const int rowh = (lane & 16) >> 1;
    #pragma unroll
    for (int r = 0; r < 8; ++r) {
      float mx = fmaxf(s[0][r], s[1][r]);
      #pragma unroll
      for (int w = 1; w < 16; w <<= 1) mx = fmaxf(mx, __shfl_xor(mx, w, 32));
      float mn   = fmaxf(mst[r], mx);
      float corr = __expf(mst[r] - mn);
      float p0   = __expf(s[0][r] - mn);
      float p1   = __expf(s[1][r] - mn);
      float rs   = p0 + p1;
      #pragma unroll
      for (int w = 1; w < 16; w <<= 1) rs += __shfl_xor(rs, w, 32);
      lst[r] = lst[r] * corr + rs;
      mst[r] = mn;
      #pragma unroll
      for (int t = 0; t < 6; ++t) oacc[t][r] *= corr;
      sP[(rowh + r) * 32 + (lane & 15)]      = p0;
      sP[(rowh + r) * 32 + 16 + (lane & 15)] = p1;
    }
    __syncthreads();

    // P (16x32) @ V (32x96)
    BF16x16 pf;
    #pragma unroll
    for (int v = 0; v < 8; ++v) {
      int kk = ((v & 4) << 2) + kbA + ((v & 3) << 1);
      pf.u[v] = packbf2(sP[m * 32 + kk], sP[m * 32 + kk + 1]);
    }
    #pragma unroll
    for (int t = 0; t < 6; ++t) {
      int col = t * 16 + (lane & 15);
      BF16x16 vf;
      #pragma unroll
      for (int v = 0; v < 8; ++v) {
        int kk = ((v & 4) << 2) + kbA + ((v & 3) << 1);
        vf.u[v] = packbf2(sV[kk * 96 + col], sV[(kk + 1) * 96 + col]);
      }
      oacc[t] = __builtin_amdgcn_wmma_f32_16x16x32_bf16(
          false, pf.v, false, vf.v, (short)0, oacc[t], false, false);
    }
    __syncthreads();
  }

  const int rowh = (lane & 16) >> 1;
  #pragma unroll
  for (int t = 0; t < 6; ++t)
    #pragma unroll
    for (int r = 0; r < 8; ++r) {
      int gm = q0 + rowh + r;
      if (gm < SQ)
        O[(size_t)gm * 768 + hoff + t * 16 + (lane & 15)] = oacc[t][r] / lst[r];
    }
}

// ---------------------------------------------------------------------------
// Selective scan: lane = (channel pair, state). 16 states per half-wave,
// recurrence in registers, y_t via 4-step shfl_xor reduction. Gating fused.
// ---------------------------------------------------------------------------
__global__ __launch_bounds__(256)
void mamba_scan(const float* __restrict__ DT,    // S x 1536 (softplus applied)
                const float* __restrict__ XC,    // S x 1536 (silu(conv))
                const float* __restrict__ XDBL,  // S x 80   (B @48, C @64)
                const float* __restrict__ XZ,    // S x 3072 (z at col 1536+)
                const float* __restrict__ A_log, // 1536 x 16
                const float* __restrict__ Dp,    // 1536
                float* __restrict__ Y,           // S x 1536
                int S)
{
  const int lane = threadIdx.x & 31;
  const int wid  = threadIdx.x >> 5;
  const int s    = lane & 15;
  const int d    = blockIdx.x * 16 + (wid << 1) + (lane >> 4);

  const float A  = -__expf(A_log[d * 16 + s]);
  const float Dd = Dp[d];
  float h = 0.f;

  for (int t = 0; t < S; ++t) {
    float dt = DT[t * 1536 + d];
    float xm = XC[t * 1536 + d];
    float Bv = XDBL[t * 80 + 48 + s];
    float Cv = XDBL[t * 80 + 64 + s];
    h = __expf(dt * A) * h + dt * xm * Bv;
    float p = h * Cv;
    #pragma unroll
    for (int w = 1; w < 16; w <<= 1) p += __shfl_xor(p, w, 32);
    if (s == 0) {
      float z = XZ[t * 3072 + 1536 + d];
      Y[t * 1536 + d] = (p + xm * Dd) * siluf(z);
    }
  }
}

// ---------------------------------------------------------------------------
// RMSNorm (optional residual-add input, optional per-row scale, optional
// scaled-residual output). One block per row.
// ---------------------------------------------------------------------------
__global__ __launch_bounds__(256)
void rmsnorm_kernel(const float* __restrict__ X,
                    const float* __restrict__ Add,       // may be null
                    const float* __restrict__ rowscale,  // may be null
                    int rs_stride, int rs_off,
                    const float* __restrict__ w,
                    float* __restrict__ Yn,
                    float* __restrict__ Res,             // may be null
                    int D)
{
  __shared__ float red[8];
  __shared__ float sinv;
  const int row = blockIdx.x;
  const float sc = rowscale ? rowscale[row * rs_stride + rs_off] : 1.f;
  const float* xr = X + (size_t)row * D;

  float ss = 0.f;
  for (int i = threadIdx.x; i < D; i += 256) {
    float v = xr[i] * sc;
    if (Add) v += Add[(size_t)row * D + i];
    ss += v * v;
  }
  #pragma unroll
  for (int m = 16; m >= 1; m >>= 1) ss += __shfl_xor(ss, m, 32);
  if ((threadIdx.x & 31) == 0) red[threadIdx.x >> 5] = ss;
  __syncthreads();
  if (threadIdx.x == 0) {
    float t = 0.f;
    for (int i = 0; i < 8; ++i) t += red[i];
    sinv = rsqrtf(t / (float)D + 1e-5f);
  }
  __syncthreads();
  const float inv = sinv;
  for (int i = threadIdx.x; i < D; i += 256) {
    float v = xr[i] * sc;
    if (Add) v += Add[(size_t)row * D + i];
    Yn[(size_t)row * D + i] = v * inv * w[i];
    if (Res) Res[(size_t)row * D + i] = v;
  }
}

// ---------------------------------------------------------------------------
// Small helpers
// ---------------------------------------------------------------------------
__global__ __launch_bounds__(256)
void embed_kernel(const float* __restrict__ E, const int* __restrict__ ids,
                  float* __restrict__ X)
{
  int s = blockIdx.x;
  size_t base = (size_t)ids[s] * 768;
  for (int i = threadIdx.x; i < 768; i += 256)
    X[(size_t)s * 768 + i] = E[base + i];
}

__global__ __launch_bounds__(256)
void router_kernel(const float* __restrict__ X,
                   const float* __restrict__ W1, const float* __restrict__ B1,
                   const float* __restrict__ W2, const float* __restrict__ B2,
                   const float* __restrict__ U,
                   float* __restrict__ RW)
{
  __shared__ float sx[768];
  __shared__ float sh[384];
  __shared__ float sl[8];
  const int row = blockIdx.x;
  for (int i = threadIdx.x; i < 768; i += 256) sx[i] = X[(size_t)row * 768 + i];
  __syncthreads();
  for (int j = threadIdx.x; j < 384; j += 256) {
    float acc = B1[j];
    const float* wr = W1 + (size_t)j * 768;
    for (int k = 0; k < 768; ++k) acc += sx[k] * wr[k];
    sh[j] = siluf(acc);
  }
  __syncthreads();
  if (threadIdx.x < 8) {
    float acc = B2[threadIdx.x];
    const float* wr = W2 + (size_t)threadIdx.x * 384;
    for (int k = 0; k < 384; ++k) acc += sh[k] * wr[k];
    sl[threadIdx.x] = acc;
  }
  __syncthreads();
  if (threadIdx.x == 0) {
    float mx = -1e30f, p[8];
    for (int i = 0; i < 8; ++i) mx = fmaxf(mx, sl[i]);
    float sum = 0.f;
    for (int i = 0; i < 8; ++i) { p[i] = __expf(sl[i] - mx); sum += p[i]; }
    for (int i = 0; i < 8; ++i) {
      float prob = p[i] / sum;
      float u = U[row * 8 + i] * (1.f - 2e-6f) + 1e-6f;
      p[i] = prob + (-logf(-logf(u)));
    }
    mx = -1e30f;
    for (int i = 0; i < 8; ++i) mx = fmaxf(mx, p[i]);
    sum = 0.f;
    for (int i = 0; i < 8; ++i) { p[i] = __expf(p[i] - mx); sum += p[i]; }
    for (int i = 0; i < 8; ++i) RW[row * 8 + i] = p[i] / sum;
  }
}

__global__ __launch_bounds__(256)
void conv_silu(const float* __restrict__ XZ,   // S x 3072 (xm = cols [0,1536))
               const float* __restrict__ CW,   // 1536 x 4
               const float* __restrict__ CB,   // 1536
               float* __restrict__ XC, int S)
{
  int idx = blockIdx.x * 256 + threadIdx.x;
  if (idx >= S * 1536) return;
  int t = idx / 1536, d = idx - t * 1536;
  float acc = CB[d];
  #pragma unroll
  for (int k = 0; k < 4; ++k) {
    int tt = t + k - 3;
    if (tt >= 0) acc += XZ[(size_t)tt * 3072 + d] * CW[d * 4 + k];
  }
  XC[idx] = siluf(acc);
}

__global__ __launch_bounds__(256)
void mix_initial(const float* __restrict__ ENC,  // (512*8) x 768 (row = s*8+i)
                 const float* __restrict__ RW,   // 512 x 8
                 float* __restrict__ OUT)        // 512 x 768
{
  int idx = blockIdx.x * 256 + threadIdx.x;
  if (idx >= 512 * 768) return;
  int s = idx / 768, d = idx - s * 768;
  float acc = 0.f;
  #pragma unroll
  for (int i = 0; i < 8; ++i)
    acc += RW[s * 8 + i] * ENC[(size_t)(s * 8 + i) * 768 + d];
  OUT[idx] = acc;
}

// ---------------------------------------------------------------------------
// Host orchestration
// ---------------------------------------------------------------------------
static void launch_gemm(hipStream_t st, const float* A, int lda, const float* W,
                        const float* bias, const float* res, int ldr,
                        float* C, int ldc, int M, int N, int K, int act)
{
  dim3 g((N + BN - 1) / BN, (M + BM - 1) / BM), b(256);
  if (act == 2)
    gemm_bf16_wmma<2, true,  false><<<g, b, 0, st>>>(A, lda, W, bias, nullptr, 0, C, ldc, M, N, K);
  else if (res)
    gemm_bf16_wmma<0, false, true ><<<g, b, 0, st>>>(A, lda, W, nullptr, res, ldr, C, ldc, M, N, K);
  else if (bias)
    gemm_bf16_wmma<0, true,  false><<<g, b, 0, st>>>(A, lda, W, bias, nullptr, 0, C, ldc, M, N, K);
  else
    gemm_bf16_wmma<0, false, false><<<g, b, 0, st>>>(A, lda, W, nullptr, nullptr, 0, C, ldc, M, N, K);
}

struct MambaP {
  const float *norm_w, *in_proj, *conv_w, *conv_b, *x_proj,
              *dt_proj, *dt_b, *A_log, *D, *out_proj;
};
static MambaP mk_mamba(void* const* d_in, int b) {
  MambaP p;
  p.norm_w  = (const float*)d_in[b + 0];
  p.in_proj = (const float*)d_in[b + 1];
  p.conv_w  = (const float*)d_in[b + 2];
  p.conv_b  = (const float*)d_in[b + 3];
  p.x_proj  = (const float*)d_in[b + 4];
  p.dt_proj = (const float*)d_in[b + 5];
  p.dt_b    = (const float*)d_in[b + 6];
  p.A_log   = (const float*)d_in[b + 7];
  p.D       = (const float*)d_in[b + 8];
  p.out_proj= (const float*)d_in[b + 9];
  return p;
}

struct Bufs { float *XN, *RES, *XZ, *XC, *XDBL, *DT, *Y; };

static void run_mamba(hipStream_t st, const MambaP& P,
                      const float* xin, const float* rw, int rw_off,
                      float* Cout, int ldc_out, const Bufs& B)
{
  // 1. normed input (and scaled residual for encoder passes)
  rmsnorm_kernel<<<512, 256, 0, st>>>(xin, nullptr, rw, 8, rw_off,
                                      P.norm_w, B.XN, rw ? B.RES : nullptr, 768);
  const float* res = rw ? (const float*)B.RES : xin;
  // 2. in_proj: 512x768 @ (3072x768)^T
  launch_gemm(st, B.XN, 768, P.in_proj, nullptr, nullptr, 0, B.XZ, 3072, 512, 3072, 768, 0);
  // 3. depthwise causal conv + silu
  conv_silu<<<(512 * 1536 + 255) / 256, 256, 0, st>>>(B.XZ, P.conv_w, P.conv_b, B.XC, 512);
  // 4. x_proj: 512x1536 @ (80x1536)^T
  launch_gemm(st, B.XC, 1536, P.x_proj, nullptr, nullptr, 0, B.XDBL, 80, 512, 80, 1536, 0);
  // 5. dt = softplus(dt_in @ dt_proj^T + dt_b), dt_in = XDBL[:, :48]
  launch_gemm(st, B.XDBL, 80, P.dt_proj, P.dt_b, nullptr, 0, B.DT, 1536, 512, 1536, 48, 2);
  // 6. selective scan + D skip + z gating
  mamba_scan<<<96, 256, 0, st>>>(B.DT, B.XC, B.XDBL, B.XZ, P.A_log, P.D, B.Y, 512);
  // 7. out_proj + residual
  launch_gemm(st, B.Y, 1536, P.out_proj, nullptr, res, 768, Cout, ldc_out, 512, 768, 1536, 0);
}

static float* wsalloc(char*& p, size_t nfloats) {
  float* r = (float*)p;
  p += ((nfloats * sizeof(float) + 255) / 256) * 256;
  return r;
}

extern "C" void kernel_launch(void* const* d_in, const int* in_sizes, int n_in,
                              void* d_out, int out_size, void* d_ws, size_t ws_size,
                              hipStream_t stream)
{
  (void)in_sizes; (void)n_in; (void)out_size; (void)ws_size;
  // Flat input order (params dict insertion order, then input_ids, gumbel_u):
  const float* embed  = (const float*)d_in[0];
  const float* r_w1   = (const float*)d_in[1];
  const float* r_b1   = (const float*)d_in[2];
  const float* r_w2   = (const float*)d_in[3];
  const float* r_b2   = (const float*)d_in[4];
  MambaP enc          = mk_mamba(d_in, 5);
  const float* agg_inw  = (const float*)d_in[15];   // 2304 x 768
  const float* agg_inb  = (const float*)d_in[16];
  const float* agg_outw = (const float*)d_in[17];
  const float* agg_outb = (const float*)d_in[18];
  const float* agg_pw   = (const float*)d_in[19];
  const float* agg_pb   = (const float*)d_in[20];
  const float* agg_nw   = (const float*)d_in[21];
  MambaP dec[4] = { mk_mamba(d_in, 22), mk_mamba(d_in, 32),
                    mk_mamba(d_in, 42), mk_mamba(d_in, 52) };
  const float* norm_f  = (const float*)d_in[62];
  const float* lm_head = (const float*)d_in[63];
  const int*   ids     = (const int*)d_in[64];
  const float* gum     = (const float*)d_in[65];
  float* out = (float*)d_out;

  char* p = (char*)d_ws;
  float* X    = wsalloc(p, 512 * 768);
  float* RW   = wsalloc(p, 512 * 8);
  Bufs B;
  B.XN   = wsalloc(p, 512 * 768);
  B.RES  = wsalloc(p, 512 * 768);
  B.XZ   = wsalloc(p, 512 * 3072);
  B.XC   = wsalloc(p, 512 * 1536);
  B.XDBL = wsalloc(p, 512 * 80);
  B.DT   = wsalloc(p, 512 * 1536);
  B.Y    = wsalloc(p, 512 * 1536);
  float* ENC  = wsalloc(p, 8 * 512 * 768);    // row = s*8+i (== seq layout)
  float* INI  = wsalloc(p, 512 * 768);
  float* QH   = wsalloc(p, 512 * 768);
  float* KH   = wsalloc(p, 4096 * 768);
  float* VH   = wsalloc(p, 4096 * 768);
  float* OA   = wsalloc(p, 512 * 768);
  float* ATT  = wsalloc(p, 512 * 768);
  float* OUTB = wsalloc(p, 512 * 768);
  float* XA   = wsalloc(p, 512 * 768);
  float* XB   = wsalloc(p, 512 * 768);

  // 1) embedding + router (Gumbel softmax weights)
  embed_kernel<<<512, 256, 0, stream>>>(embed, ids, X);
  router_kernel<<<512, 256, 0, stream>>>(X, r_w1, r_b1, r_w2, r_b2, gum, RW);

  // 2) 8 encoder passes with shared weights; outputs interleaved (s*8+i)
  for (int i = 0; i < 8; ++i)
    run_mamba(stream, enc, X, RW, i, ENC + (size_t)i * 768, 8 * 768, B);

  // 3) router-weighted mix -> initial (queries); ENC is already `seq`
  mix_initial<<<(512 * 768 + 255) / 256, 256, 0, stream>>>(ENC, RW, INI);

  // 4) cross attention: q from initial (512), kv from seq (4096)
  launch_gemm(stream, INI, 768, agg_inw,                 agg_inb,        nullptr, 0, QH, 768, 512,  768, 768, 0);
  launch_gemm(stream, ENC, 768, agg_inw + 768 * 768,     agg_inb + 768,  nullptr, 0, KH, 768, 4096, 768, 768, 0);
  launch_gemm(stream, ENC, 768, agg_inw + 2 * 768 * 768, agg_inb + 1536, nullptr, 0, VH, 768, 4096, 768, 768, 0);
  attn_flash<<<dim3(32, 8), 32, 0, stream>>>(QH, KH, VH, OA, 512, 4096);
  launch_gemm(stream, OA,  768, agg_outw, agg_outb, nullptr, 0, ATT,  768, 512, 768, 768, 0);
  launch_gemm(stream, ATT, 768, agg_pw,   agg_pb,   nullptr, 0, OUTB, 768, 512, 768, 768, 0);

  // 5) x = rmsnorm(out + initial)
  rmsnorm_kernel<<<512, 256, 0, stream>>>(OUTB, INI, nullptr, 0, 0, agg_nw, XA, nullptr, 768);

  // 6) decoder mamba stack
  float* xcur = XA; float* xnext = XB;
  for (int l = 0; l < 4; ++l) {
    run_mamba(stream, dec[l], xcur, nullptr, 0, xnext, 768, B);
    float* t = xcur; xcur = xnext; xnext = t;
  }

  // 7) final norm + lm_head -> logits (512 x 32000)
  rmsnorm_kernel<<<512, 256, 0, stream>>>(xcur, nullptr, nullptr, 0, 0, norm_f, B.XN, nullptr, 768);
  launch_gemm(stream, B.XN, 768, lm_head, nullptr, nullptr, 0, out, 32000, 512, 32000, 768, 0);
}
